// Decoder_74328704024734
// MI455X (gfx1250) — matrix operand order repaired and checked by reading, not compiled, not run on hardware
//
#include <hip/hip_runtime.h>
#include <hip/hip_bf16.h>

typedef __attribute__((ext_vector_type(16))) _Float16 v16h;
typedef __attribute__((ext_vector_type(8)))  _Float16 v8h;
typedef __attribute__((ext_vector_type(8)))  float    v8f;

#define N_PHASES 256
#define N_PROV   81
#define H_ENC    120
#define EMB      80
#define LSTM_D   200
#define VOCAB    13042
#define N_LOC    32
#define GO_IDX   1
#define ALPHA    0.2f

#define KPAD      224           // 200 -> 7 k-tiles of 32
#define KT        7
#define GATE_NT   13            // 200 -> 13 n-tiles of 16 (pad 208)
#define DENSE_NT  816           // 13042 -> 816 n-tiles of 16 (pad 13056)
#define DENSE_TPW 4             // n-tiles per wave in dense kernel
#define FRAG      512           // halves per packed 32x16 B tile (32 lanes x 16)

__device__ __forceinline__ float sigf(float x)  { return 1.0f / (1.0f + __expf(-x)); }
__device__ __forceinline__ float lreluf(float x){ return x >= 0.0f ? x : ALPHA * x; }

// ---- A fragment: 16x32 f16 tile from padded row-major buffer (ld=KPAD) ----
// ISA layout: lane<16 holds K {k0..k0+7, k0+16..k0+23}; lane>=16 holds +8.
// Two aligned 128-bit loads per lane.
__device__ __forceinline__ v16h frag_A(const _Float16* __restrict__ base,
                                       int row, int k0, int lane) {
    const int hi = lane >> 4;
    const _Float16* p = base + (size_t)row * KPAD + k0 + hi * 8;
    v8h lo  = *(const v8h*)(p);
    v8h hi8 = *(const v8h*)(p + 16);
    v16h f;
#pragma unroll
    for (int e = 0; e < 8; ++e) { f[e] = lo[e]; f[8 + e] = hi8[e]; }
    return f;
}

// ---- B fragment: packed [tile][lane][16 halves]; one 32-byte load ----
__device__ __forceinline__ v16h frag_B(const _Float16* __restrict__ pack,
                                       int tile, int lane) {
    return *(const v16h*)(pack + ((size_t)tile * 32 + lane) * 16);
}

// ---- packing kernels (run once per call) -------------------------------
// Gate weights: src (200 x 800 f32) -> dst packed [4][13][7][32][16] f16, zero-padded
__global__ void pack_gate_kernel(const float* __restrict__ src,
                                 _Float16* __restrict__ dst) {
    const int total = 4 * GATE_NT * KT * FRAG;
    for (int i = blockIdx.x * blockDim.x + threadIdx.x; i < total;
         i += gridDim.x * blockDim.x) {
        int e    = i & 15;
        int lane = (i >> 4) & 31;
        int kt   = (i >> 9) % KT;
        int nt   = (i / (FRAG * KT)) % GATE_NT;
        int g    = i / (FRAG * KT * GATE_NT);
        int k    = kt * 32 + (e & 7) + (lane >> 4) * 8 + (e >> 3) * 16;
        int nl   = nt * 16 + (lane & 15);
        float v  = (k < LSTM_D && nl < LSTM_D)
                 ? src[(size_t)k * (4 * LSTM_D) + g * LSTM_D + nl] : 0.0f;
        dst[i] = (_Float16)v;
    }
}

// Dense weights: src (200 x 13042 f32) -> dst packed [816][7][32][16] f16
__global__ void pack_dense_kernel(const float* __restrict__ src,
                                  _Float16* __restrict__ dst) {
    const int total = DENSE_NT * KT * FRAG;
    for (int i = blockIdx.x * blockDim.x + threadIdx.x; i < total;
         i += gridDim.x * blockDim.x) {
        int e    = i & 15;
        int lane = (i >> 4) & 31;
        int kt   = (i >> 9) % KT;
        int nt   = i / (FRAG * KT);
        int k    = kt * 32 + (e & 7) + (lane >> 4) * 8 + (e >> 3) * 16;
        int n    = nt * 16 + (lane & 15);
        float v  = (k < LSTM_D && n < VOCAB) ? src[(size_t)k * VOCAB + n] : 0.0f;
        dst[i] = (_Float16)v;
    }
}

// Per-step state matrices: states[s] = [0_80 | h_enc[:,loc_s,:]] padded to 224, f16
__global__ void pack_states_kernel(const float* __restrict__ h_enc,
                                   const int* __restrict__ locations,
                                   _Float16* __restrict__ states) {
    const int total = N_LOC * N_PHASES * KPAD;
    for (int i = blockIdx.x * blockDim.x + threadIdx.x; i < total;
         i += gridDim.x * blockDim.x) {
        int k   = i % KPAD;
        int row = (i / KPAD) % N_PHASES;
        int s   = i / (KPAD * N_PHASES);
        float v = 0.0f;
        if (k >= EMB && k < LSTM_D) {
            int loc = locations[s];
            v = h_enc[((size_t)row * N_PROV + loc) * H_ENC + (k - EMB)];
        }
        states[i] = (_Float16)v;
    }
}

// h0 = [emb[GO] broadcast | zeros], both h buffers fully zero-padded
__global__ void init_h_kernel(const float* __restrict__ emb_table,
                              _Float16* __restrict__ h0,
                              _Float16* __restrict__ h1) {
    const int total = N_PHASES * KPAD;
    for (int i = blockIdx.x * blockDim.x + threadIdx.x; i < total;
         i += gridDim.x * blockDim.x) {
        int col = i % KPAD;
        h0[i] = (_Float16)(col < EMB ? emb_table[GO_IDX * EMB + col] : 0.0f);
        h1[i] = (_Float16)0.0f;
    }
}

// ---- LSTM step: one wave computes a 16x16 tile of h_new for all 4 gates --
// grid = (13, 16), block = 32. 4 independent WMMA accumulation chains.
__global__ void lstm_step_kernel(const _Float16* __restrict__ WxP,  // packed
                                 const _Float16* __restrict__ WhP,  // packed
                                 const float* __restrict__ lstm_b,  // 800
                                 const _Float16* __restrict__ state_s, // 256 x 224 f16
                                 const _Float16* __restrict__ h_prev,  // 256 x 224 f16
                                 _Float16* __restrict__ h_out) {       // 256 x 224 f16
    const int lane = threadIdx.x & 31;
    const int hi   = lane >> 4;
    const int m0   = blockIdx.y * 16;
    const int nt   = blockIdx.x;           // 0..12
    const int row  = m0 + (lane & 15);
    const int col  = nt * 16 + (lane & 15);
    const bool cok = col < LSTM_D;

    v8f acc[4];
#pragma unroll
    for (int g = 0; g < 4; ++g) acc[g] = (v8f){};

#pragma unroll
    for (int kt = 0; kt < KT; ++kt) {
        v16h a_prev = frag_A(h_prev, row, kt * 32, lane);
        v16h a_st   = frag_A(state_s, row, kt * 32, lane);
#pragma unroll
        for (int g = 0; g < 4; ++g) {
            int tile = (g * GATE_NT + nt) * KT + kt;
            v16h bx = frag_B(WxP, tile, lane);
            acc[g] = __builtin_amdgcn_wmma_f32_16x16x32_f16(
                false, a_prev, false, bx, (short)0, acc[g], false, false);
            v16h bh = frag_B(WhP, tile, lane);
            acc[g] = __builtin_amdgcn_wmma_f32_16x16x32_f16(
                false, a_st, false, bh, (short)0, acc[g], false, false);
        }
    }

    const int bcol = cok ? col : 0;
    float bI = lstm_b[0 * LSTM_D + bcol];
    float bF = lstm_b[1 * LSTM_D + bcol];
    float bG = lstm_b[2 * LSTM_D + bcol];
    float bO = lstm_b[3 * LSTM_D + bcol];

#pragma unroll
    for (int r = 0; r < 8; ++r) {
        int grow = m0 + r + hi * 8;
        float zi = acc[0][r] + bI;
        float zf = acc[1][r] + bF;
        float zg = acc[2][r] + bG;
        float zo = acc[3][r] + bO;
        float st = (float)state_s[(size_t)grow * KPAD + (cok ? col : 0)];
        float cn = sigf(zf) * st + sigf(zi) * lreluf(zg);
        float hn = sigf(zo) * lreluf(cn);
        if (cok) h_out[(size_t)grow * KPAD + col] = (_Float16)hn;
    }
}

// ---- dense logits: one wave per 16x64 tile (4 independent acc chains) ---
// grid = (204, 16), block = 32
__global__ void dense_logits_kernel(const _Float16* __restrict__ h,   // 256 x 224 f16
                                    const _Float16* __restrict__ WdP, // packed
                                    const float* __restrict__ dense_b,
                                    float* __restrict__ out) {        // 256 x 13042
    const int lane = threadIdx.x & 31;
    const int hi   = lane >> 4;
    const int m0   = blockIdx.y * 16;
    const int nt0  = blockIdx.x * DENSE_TPW;
    const int row  = m0 + (lane & 15);

    v8f acc[DENSE_TPW];
#pragma unroll
    for (int t = 0; t < DENSE_TPW; ++t) acc[t] = (v8f){};

#pragma unroll
    for (int kt = 0; kt < KT; ++kt) {
        v16h a = frag_A(h, row, kt * 32, lane);
#pragma unroll
        for (int t = 0; t < DENSE_TPW; ++t) {
            v16h b = frag_B(WdP, (nt0 + t) * KT + kt, lane);
            acc[t] = __builtin_amdgcn_wmma_f32_16x16x32_f16(
                false, a, false, b, (short)0, acc[t], false, false);
        }
    }

#pragma unroll
    for (int t = 0; t < DENSE_TPW; ++t) {
        int col = (nt0 + t) * 16 + (lane & 15);
        if (col < VOCAB) {
            float bias = dense_b[col];
#pragma unroll
            for (int r = 0; r < 8; ++r) {
                int grow = m0 + r + hi * 8;
                out[(size_t)grow * VOCAB + col] = acc[t][r] + bias;
            }
        }
    }
}

// ---- per-row softmax (in place) + argmax -------------------------------
// grid = 256, block = 256
__global__ void softmax_argmax_kernel(float* __restrict__ probs,
                                      float* __restrict__ actions) {
    const int row = blockIdx.x;
    const int tid = threadIdx.x;
    float* p = probs + (size_t)row * VOCAB;

    __shared__ float smax[256];
    __shared__ int   sidx[256];
    __shared__ float ssum[256];

    float m = -3.402823466e38f;
    int   mi = VOCAB;
    for (int i = tid; i < VOCAB; i += 256) {
        float v = p[i];
        if (v > m) { m = v; mi = i; }
    }
    smax[tid] = m; sidx[tid] = mi;
    __syncthreads();
    for (int s = 128; s > 0; s >>= 1) {
        if (tid < s) {
            float ov = smax[tid + s]; int oi = sidx[tid + s];
            if (ov > smax[tid] || (ov == smax[tid] && oi < sidx[tid])) {
                smax[tid] = ov; sidx[tid] = oi;
            }
        }
        __syncthreads();
    }
    const float rmax = smax[0];
    const int   amax = sidx[0];

    float sum = 0.0f;
    for (int i = tid; i < VOCAB; i += 256) sum += __expf(p[i] - rmax);
    ssum[tid] = sum;
    __syncthreads();
    for (int s = 128; s > 0; s >>= 1) {
        if (tid < s) ssum[tid] += ssum[tid + s];
        __syncthreads();
    }
    const float inv = 1.0f / ssum[0];

    for (int i = tid; i < VOCAB; i += 256) p[i] = __expf(p[i] - rmax) * inv;

    if (tid == 0) actions[row] = (float)amax;
}

// ---- launch ------------------------------------------------------------
extern "C" void kernel_launch(void* const* d_in, const int* in_sizes, int n_in,
                              void* d_out, int out_size, void* d_ws, size_t ws_size,
                              hipStream_t stream) {
    const float* h_enc     = (const float*)d_in[0];
    const float* emb_table = (const float*)d_in[1];
    const float* lstm_Wx   = (const float*)d_in[2];
    const float* lstm_Wh   = (const float*)d_in[3];
    const float* lstm_b    = (const float*)d_in[4];
    const float* dense_W   = (const float*)d_in[5];
    const float* dense_b   = (const float*)d_in[6];
    const int*   locations = (const int*)d_in[7];

    float* out_actions = (float*)d_out;                  // 32*256
    float* out_probs   = out_actions + N_LOC * N_PHASES; // 32*256*13042

    // ws layout (bytes, 256-aligned)
    char* ws = (char*)d_ws;
    _Float16* WxP    = (_Float16*)(ws + 0);         // 4*13*7*512*2 = 372736
    _Float16* WhP    = (_Float16*)(ws + 372736);    // 372736
    _Float16* WdP    = (_Float16*)(ws + 745472);    // 816*7*512*2 = 5849088
    _Float16* states = (_Float16*)(ws + 6594560);   // 32*256*224*2 = 3670016
    _Float16* h0     = (_Float16*)(ws + 10264576);  // 256*224*2 = 114688
    _Float16* h1     = (_Float16*)(ws + 10379264);  // 114688
    _Float16* hbuf[2] = { h0, h1 };

    pack_gate_kernel <<<728, 256, 0, stream>>>(lstm_Wx, WxP);   // 186368 elems
    pack_gate_kernel <<<728, 256, 0, stream>>>(lstm_Wh, WhP);
    pack_dense_kernel<<<2856, 256, 0, stream>>>(dense_W, WdP);  // 2924544 elems
    pack_states_kernel<<<1792, 256, 0, stream>>>(h_enc, locations, states); // 1835008
    init_h_kernel    <<<224, 256, 0, stream>>>(emb_table, h0, h1);

    const dim3 lstm_grid(GATE_NT, 16);
    const dim3 dense_grid(DENSE_NT / DENSE_TPW, 16);   // (204, 16)

    for (int s = 0; s < N_LOC; ++s) {
        const _Float16* hp = hbuf[s & 1];
        _Float16* hn       = hbuf[(s + 1) & 1];
        const _Float16* st = states + (size_t)s * N_PHASES * KPAD;
        float* probs_s     = out_probs + (size_t)s * N_PHASES * VOCAB;

        lstm_step_kernel<<<lstm_grid, 32, 0, stream>>>(
            WxP, WhP, lstm_b, st, hp, hn);
        dense_logits_kernel<<<dense_grid, 32, 0, stream>>>(
            hn, WdP, dense_b, probs_s);
        softmax_argmax_kernel<<<N_PHASES, 256, 0, stream>>>(
            probs_s, out_actions + (size_t)s * N_PHASES);
    }
}
